// MyQwenAttention_14199161881025
// MI455X (gfx1250) — matrix operand order repaired and checked by reading, not compiled, not run on hardware
//
#include <hip/hip_runtime.h>
#include <hip/hip_bf16.h>

// ---- problem constants (from reference) ----
#define Bc 2
#define Sc 2048
#define QHc 14
#define KVHc 2
#define Dc 64
#define HIDc 896
#define GROUPc 7          // QHc / KVHc
#define MTOK (Bc * Sc)    // 4096 tokens

typedef __attribute__((ext_vector_type(16))) _Float16 v16h;
typedef __attribute__((ext_vector_type(8)))  float    v8f;

// A-fragment (16x32 f16) K index for element e, lane-half g (contiguous: k = 16g + e, matches B fills)
__device__ __forceinline__ int a_kidx(int e, int g) {
  return 16 * g + e;
}

__device__ __forceinline__ v8f wmma_f16(v16h a, v16h b, v8f c) {
  return __builtin_amdgcn_wmma_f32_16x16x32_f16(false, a, false, b, (short)0, c,
                                                false, false);
}

// -------------------------------------------------------------------------
// Tiled WMMA GEMM: C[M,N] = A[M,K] * W[N,K]^T + bias   (C f32, A f32 or f16)
// Block: 256 threads = 8 waves, macro tile 32(M) x 64(N), K-step 32.
// -------------------------------------------------------------------------
template <typename AT>
__global__ void __launch_bounds__(256)
gemm_wmma_kernel(const AT* __restrict__ A, const float* __restrict__ W,
                 const float* __restrict__ bias, float* __restrict__ C,
                 int M, int N, int K) {
  __shared__ _Float16 Atile[32][34];   // +2 pad: conflict-free column gathers
  __shared__ _Float16 Btile[32][64];   // [k][n], row gathers are conflict-free

  const int t   = threadIdx.x;
  const int lane = t & 31;
  const int wid  = t >> 5;
  const int wm   = wid >> 2;        // 0..1
  const int wn   = wid & 3;         // 0..3
  const int g    = lane >> 4;
  const int l15  = lane & 15;
  const int m0   = blockIdx.y * 32;
  const int n0   = blockIdx.x * 64;

  v8f c = {};

  for (int kk = 0; kk < K; kk += 32) {
    __syncthreads();
    // stage A tile 32x32 (convert to f16)
#pragma unroll
    for (int i = 0; i < 4; ++i) {
      int e = t + i * 256;
      int r = e >> 5, cc = e & 31;
      Atile[r][cc] = (_Float16)A[(size_t)(m0 + r) * K + kk + cc];
    }
    // stage B tile: W rows n0..n0+63, cols kk..kk+31, transposed -> [k][n]
#pragma unroll
    for (int i = 0; i < 8; ++i) {
      int e = t + i * 256;
      int n = e >> 5, k = e & 31;
      Btile[k][n] = (_Float16)W[(size_t)(n0 + n) * K + kk + k];
    }
    __syncthreads();

    v16h a, b;
#pragma unroll
    for (int e = 0; e < 16; ++e) {
      a[e] = Atile[wm * 16 + l15][a_kidx(e, g)];
      b[e] = Btile[16 * g + e][wn * 16 + l15];
    }
    c = wmma_f16(a, b, c);
  }

  // epilogue: row m = m0 + wm*16 + r + 8*g ; col n = n0 + wn*16 + l15
  const int n  = n0 + wn * 16 + l15;
  const int mb = m0 + wm * 16 + 8 * g;
  const float bn = bias ? bias[n] : 0.0f;
#pragma unroll
  for (int r = 0; r < 8; ++r) {
    C[(size_t)(mb + r) * N + n] = c[r] + bn;
  }
}

// -------------------------------------------------------------------------
// RoPE + f16 conversion into head-major layouts.
// q_raw: [tok][QH*64], k_raw/v_raw: [tok][KVH*64]
// Qh: [b][qh][s][64] (scaled by 1/sqrt(D)), Kh/Vh: [b][kvh][s][64]
// -------------------------------------------------------------------------
__global__ void __launch_bounds__(256)
rope_convert_kernel(const float* __restrict__ q_raw,
                    const float* __restrict__ k_raw,
                    const float* __restrict__ v_raw,
                    const float* __restrict__ cosb,
                    const float* __restrict__ sinb,
                    _Float16* __restrict__ Qh, _Float16* __restrict__ Kh,
                    _Float16* __restrict__ Vh) {
  const int NQ  = Bc * Sc * QHc * Dc;
  const int NKV = Bc * Sc * KVHc * Dc;
  int tid = blockIdx.x * blockDim.x + threadIdx.x;

  if (tid < NQ) {
    int d = tid & 63;
    int tmp = tid >> 6;
    int h = tmp % QHc;
    int tok = tmp / QHc;
    int s = tok % Sc;
    int b = tok / Sc;
    float cv = cosb[s * Dc + d];
    float sv = sinb[s * Dc + d];
    float x = q_raw[tid];
    float rot = (d < 32) ? -q_raw[tid + 32] : q_raw[tid - 32];
    float val = (x * cv + rot * sv) * 0.125f;  // fold 1/sqrt(64)
    Qh[(((size_t)b * QHc + h) * Sc + s) * Dc + d] = (_Float16)val;
    return;
  }
  int tid2 = tid - NQ;
  if (tid2 < NKV) {
    int d = tid2 & 63;
    int tmp = tid2 >> 6;
    int h = tmp % KVHc;
    int tok = tmp / KVHc;
    int s = tok % Sc;
    int b = tok / Sc;
    float cv = cosb[s * Dc + d];
    float sv = sinb[s * Dc + d];
    float x = k_raw[tid2];
    float rot = (d < 32) ? -k_raw[tid2 + 32] : k_raw[tid2 - 32];
    Kh[(((size_t)b * KVHc + h) * Sc + s) * Dc + d] = (_Float16)(x * cv + rot * sv);
    return;
  }
  int tid3 = tid2 - NKV;
  if (tid3 < NKV) {
    int d = tid3 & 63;
    int tmp = tid3 >> 6;
    int h = tmp % KVHc;
    int tok = tmp / KVHc;
    int s = tok % Sc;
    int b = tok / Sc;
    Vh[(((size_t)b * KVHc + h) * Sc + s) * Dc + d] = (_Float16)v_raw[tid3];
  }
}

// -------------------------------------------------------------------------
// Flash attention (causal, GQA). Block = 128 threads = 4 waves.
// Each wave owns 16 queries; block owns 64 queries of one (b, h).
// Key tiles of 32; K/V staged in LDS, scores & P*V via WMMA f16.
// ctxh out: token-major f16 [b][s][qh*64 + d]  (feeds O-projection GEMM)
// -------------------------------------------------------------------------
__global__ void __launch_bounds__(128)
attn_kernel(const _Float16* __restrict__ Qh, const _Float16* __restrict__ Kh,
            const _Float16* __restrict__ Vh, _Float16* __restrict__ ctxh) {
  __shared__ _Float16 Kt[32][66];         // [key][d], padded
  __shared__ _Float16 Vt[32][66];
  __shared__ _Float16 Pt[4][16][34];      // per-wave P reshape buffer

  const int t = threadIdx.x;
  const int lane = t & 31;
  const int wid = t >> 5;
  const int g = lane >> 4;
  const int l15 = lane & 15;

  const int q0b = blockIdx.x * 64;
  const int h   = blockIdx.y;
  const int b   = blockIdx.z;
  const int kvh = h / GROUPc;
  const int q0w = q0b + wid * 16;

  const size_t qbase  = (((size_t)b * QHc + h) * Sc + q0w) * Dc;
  const size_t kvbase = ((size_t)b * KVHc + kvh) * Sc * Dc;

  // Q fragments (constant over key loop): d-chunks 0..31, 32..63
  v16h aQ0, aQ1;
#pragma unroll
  for (int e = 0; e < 16; ++e) {
    int kidx = a_kidx(e, g);
    aQ0[e] = Qh[qbase + (size_t)l15 * Dc + kidx];
    aQ1[e] = Qh[qbase + (size_t)l15 * Dc + 32 + kidx];
  }

  v8f acc[4] = {};
  float mrow[8], lrow[8];
#pragma unroll
  for (int r = 0; r < 8; ++r) { mrow[r] = -3.0e38f; lrow[r] = 0.0f; }

  const int nkt = q0b / 32 + 2;  // tiles covering keys 0 .. q0b+63

  for (int kt = 0; kt < nkt; ++kt) {
    const int key0 = kt * 32;
    __syncthreads();
    // cooperative K/V tile load (32 keys x 64 d)
#pragma unroll
    for (int i = 0; i < 16; ++i) {
      int e = t + i * 128;
      int ky = e >> 6, d = e & 63;
      Kt[ky][d] = Kh[kvbase + (size_t)(key0 + ky) * Dc + d];
      Vt[ky][d] = Vh[kvbase + (size_t)(key0 + ky) * Dc + d];
    }
    __syncthreads();

    if (key0 <= q0w + 15) {   // wave-uniform: EXEC stays all-ones
      // ---- scores: S = Q * K^T, two 16-key sub-tiles ----
      v16h bK;
      v8f s0 = {}, s1 = {};
#pragma unroll
      for (int e = 0; e < 16; ++e) bK[e] = Kt[l15][16 * g + e];
      s0 = wmma_f16(aQ0, bK, s0);
#pragma unroll
      for (int e = 0; e < 16; ++e) bK[e] = Kt[l15][32 + 16 * g + e];
      s0 = wmma_f16(aQ1, bK, s0);
#pragma unroll
      for (int e = 0; e < 16; ++e) bK[e] = Kt[16 + l15][16 * g + e];
      s1 = wmma_f16(aQ0, bK, s1);
#pragma unroll
      for (int e = 0; e < 16; ++e) bK[e] = Kt[16 + l15][32 + 16 * g + e];
      s1 = wmma_f16(aQ1, bK, s1);

      // ---- causal mask + online softmax (row r+8g, col l15 / 16+l15) ----
#pragma unroll
      for (int r = 0; r < 8; ++r) {
        int qp = q0w + 8 * g + r;
        float x0 = s0[r], x1 = s1[r];
        if (key0 + l15 > qp)      x0 = -3.0e38f;
        if (key0 + 16 + l15 > qp) x1 = -3.0e38f;

        float mx = fmaxf(x0, x1);
#pragma unroll
        for (int off = 1; off < 16; off <<= 1)
          mx = fmaxf(mx, __shfl_xor(mx, off, 32));
        float mnew  = fmaxf(mrow[r], mx);
        float alpha = __expf(mrow[r] - mnew);
        float p0 = __expf(x0 - mnew);
        float p1 = __expf(x1 - mnew);
        float rs = p0 + p1;
#pragma unroll
        for (int off = 1; off < 16; off <<= 1)
          rs += __shfl_xor(rs, off, 32);
        lrow[r] = lrow[r] * alpha + rs;
        mrow[r] = mnew;
#pragma unroll
        for (int dt = 0; dt < 4; ++dt) acc[dt][r] *= alpha;

        Pt[wid][8 * g + r][l15]      = (_Float16)p0;
        Pt[wid][8 * g + r][16 + l15] = (_Float16)p1;
      }

      // ---- ctx += P * V ----
      v16h aP;
#pragma unroll
      for (int e = 0; e < 16; ++e) aP[e] = Pt[wid][l15][a_kidx(e, g)];
#pragma unroll
      for (int dt = 0; dt < 4; ++dt) {
        v16h bV;
#pragma unroll
        for (int e = 0; e < 16; ++e) bV[e] = Vt[16 * g + e][dt * 16 + l15];
        acc[dt] = wmma_f16(aP, bV, acc[dt]);
      }
    }
  }

  // epilogue: normalize and write token-major f16 context
#pragma unroll
  for (int dt = 0; dt < 4; ++dt) {
#pragma unroll
    for (int r = 0; r < 8; ++r) {
      int qp = q0w + 8 * g + r;
      int d = dt * 16 + l15;
      ctxh[((size_t)b * Sc + qp) * (QHc * Dc) + h * Dc + d] =
          (_Float16)(acc[dt][r] / lrow[r]);
    }
  }
}

// -------------------------------------------------------------------------
extern "C" void kernel_launch(void* const* d_in, const int* in_sizes, int n_in,
                              void* d_out, int out_size, void* d_ws,
                              size_t ws_size, hipStream_t stream) {
  (void)in_sizes; (void)n_in; (void)out_size; (void)ws_size;

  const float* hidden = (const float*)d_in[0];
  const float* cosb   = (const float*)d_in[1];
  const float* sinb   = (const float*)d_in[2];
  const float* q_w    = (const float*)d_in[3];
  const float* q_b    = (const float*)d_in[4];
  const float* k_w    = (const float*)d_in[5];
  const float* k_b    = (const float*)d_in[6];
  const float* v_w    = (const float*)d_in[7];
  const float* v_b    = (const float*)d_in[8];
  const float* o_w    = (const float*)d_in[9];
  float* out = (float*)d_out;

  char* ws = (char*)d_ws;
  size_t off = 0;
  auto alloc = [&](size_t bytes) -> void* {
    void* p = ws + off;
    off += (bytes + 255) & ~(size_t)255;
    return p;
  };

  float* q_raw = (float*)alloc((size_t)MTOK * (QHc * Dc) * sizeof(float));
  float* k_raw = (float*)alloc((size_t)MTOK * (KVHc * Dc) * sizeof(float));
  float* v_raw = (float*)alloc((size_t)MTOK * (KVHc * Dc) * sizeof(float));
  _Float16* Qh   = (_Float16*)alloc((size_t)MTOK * (QHc * Dc) * sizeof(_Float16));
  _Float16* Kh   = (_Float16*)alloc((size_t)MTOK * (KVHc * Dc) * sizeof(_Float16));
  _Float16* Vh   = (_Float16*)alloc((size_t)MTOK * (KVHc * Dc) * sizeof(_Float16));
  _Float16* ctxh = (_Float16*)alloc((size_t)MTOK * (QHc * Dc) * sizeof(_Float16));

  // 1) QKV projections (WMMA GEMM, f32 A converted on LDS stage)
  gemm_wmma_kernel<float><<<dim3((QHc * Dc) / 64, MTOK / 32), 256, 0, stream>>>(
      hidden, q_w, q_b, q_raw, MTOK, QHc * Dc, HIDc);
  gemm_wmma_kernel<float><<<dim3((KVHc * Dc) / 64, MTOK / 32), 256, 0, stream>>>(
      hidden, k_w, k_b, k_raw, MTOK, KVHc * Dc, HIDc);
  gemm_wmma_kernel<float><<<dim3((KVHc * Dc) / 64, MTOK / 32), 256, 0, stream>>>(
      hidden, v_w, v_b, v_raw, MTOK, KVHc * Dc, HIDc);

  // 2) RoPE + f16 head-major conversion
  {
    int total = Bc * Sc * QHc * Dc + 2 * (Bc * Sc * KVHc * Dc);
    rope_convert_kernel<<<(total + 255) / 256, 256, 0, stream>>>(
        q_raw, k_raw, v_raw, cosb, sinb, Qh, Kh, Vh);
  }

  // 3) Flash attention (causal, GQA) -> f16 context, token-major
  attn_kernel<<<dim3(Sc / 64, QHc, Bc), 128, 0, stream>>>(Qh, Kh, Vh, ctxh);

  // 4) Output projection -> d_out (f32)
  gemm_wmma_kernel<_Float16><<<dim3(HIDc / 64, MTOK / 32), 256, 0, stream>>>(
      ctxh, o_w, nullptr, out, MTOK, HIDc, HIDc);
}
